// Global_Positional_Relational_Block_38405597561064
// MI455X (gfx1250) — matrix operand order repaired and checked by reading, not compiled, not run on hardware
//
#include <hip/hip_runtime.h>
#include <hip/hip_bf16.h>
#include <stdint.h>

// ---------------- problem constants ----------------
#define BB 2
#define TT 2048
#define DD 1024
#define HH 16
#define HDIM 64
#define BT (BB * TT)            // 4096 rows
#define BTD ((size_t)BT * DD)   // 4,194,304 elements
#define NC 32                   // EMA chunks
#define CL 64                   // EMA chunk length

typedef __bf16        v16bf __attribute__((ext_vector_type(16)));
typedef float         v8f   __attribute__((ext_vector_type(8)));
typedef unsigned int  u32x4 __attribute__((ext_vector_type(4)));
typedef unsigned int  u32x2 __attribute__((ext_vector_type(2)));

union Frag { v16bf v; u32x4 q[2]; };

// ---- optional gfx1250 async global->LDS path (compile-safe probe) ----
#if defined(__has_builtin)
#if __has_builtin(__builtin_amdgcn_global_load_async_to_lds_b128) && \
    __has_builtin(__builtin_amdgcn_s_wait_asynccnt)
#define USE_ASYNC_LDS 1
#endif
#endif
#ifndef USE_ASYNC_LDS
#define USE_ASYNC_LDS 0
#endif

#if USE_ASYNC_LDS
// builtin signature (from hipcc diagnostic): param 0 is
// '__attribute__((vector_size(16))) int __device__ *' (global int4*)
typedef int v4i_t __attribute__((vector_size(16)));
typedef __attribute__((address_space(1))) v4i_t* g_v4i_p;
typedef __attribute__((address_space(3))) v4i_t* l_v4i_p;
__device__ __forceinline__ void async_cp16(const void* g, void* l) {
  __builtin_amdgcn_global_load_async_to_lds_b128(
      (g_v4i_p)(unsigned long long)(uintptr_t)g, (l_v4i_p)l, 0, 0);
}
#endif

__device__ __forceinline__ unsigned short f2bf(float f) {
  unsigned int u = __float_as_uint(f);
  unsigned int r = u + 0x7FFFu + ((u >> 16) & 1u);   // round-to-nearest-even
  return (unsigned short)(r >> 16);
}

// ------- 16-lane row reductions via DPP (no LDS traffic) -------
#define DPP8_SEL(a,b,c,d,e,f,g,h) \
  ((a)|((b)<<3)|((c)<<6)|((d)<<9)|((e)<<12)|((f)<<15)|((g)<<18)|((h)<<21))

__device__ __forceinline__ float dpp_xor1(float v) {
  return __int_as_float(__builtin_amdgcn_mov_dpp8(__float_as_int(v),
                        DPP8_SEL(1,0,3,2,5,4,7,6)));
}
__device__ __forceinline__ float dpp_xor2(float v) {
  return __int_as_float(__builtin_amdgcn_mov_dpp8(__float_as_int(v),
                        DPP8_SEL(2,3,0,1,6,7,4,5)));
}
__device__ __forceinline__ float dpp_xor4(float v) {
  return __int_as_float(__builtin_amdgcn_mov_dpp8(__float_as_int(v),
                        DPP8_SEL(4,5,6,7,0,1,2,3)));
}
__device__ __forceinline__ float dpp_xor8(float v) {
  // DPP16 row_ror:8 == xor-8 within each 16-lane row (all lanes active)
  return __int_as_float(__builtin_amdgcn_update_dpp(
      0, __float_as_int(v), 0x128, 0xf, 0xf, true));
}
__device__ __forceinline__ float redmax16(float v) {
  v = fmaxf(v, dpp_xor1(v));
  v = fmaxf(v, dpp_xor2(v));
  v = fmaxf(v, dpp_xor4(v));
  v = fmaxf(v, dpp_xor8(v));
  return v;
}
__device__ __forceinline__ float redsum16(float v) {
  v += dpp_xor1(v);
  v += dpp_xor2(v);
  v += dpp_xor4(v);
  v += dpp_xor8(v);
  return v;
}

// =================================================================
// EMA chunked scan: e_t = 0.9 e_{t-1} + 0.1 x_t,  e_0 = x_0
// =================================================================
__global__ __launch_bounds__(256) void ema_phaseA(const float* __restrict__ x,
                                                  float* __restrict__ carry) {
  int i = blockIdx.x * 256 + threadIdx.x;          // B*NC*D threads
  int d = i % DD;
  int c = (i / DD) % NC;
  int b = i / (DD * NC);
  size_t base = ((size_t)b * TT + c * CL) * DD + d;
  float e = 0.f;
  if (c == 0) {
    e = x[base];
    #pragma unroll 4
    for (int t = 1; t < CL; ++t) e = 0.9f * e + 0.1f * x[base + (size_t)t * DD];
  } else {
    #pragma unroll 4
    for (int t = 0; t < CL; ++t) e = 0.9f * e + 0.1f * x[base + (size_t)t * DD];
  }
  carry[((size_t)b * NC + c) * DD + d] = e;
}

__global__ __launch_bounds__(256) void ema_phaseB(const float* __restrict__ carry,
                                                  float* __restrict__ estart) {
  int i = blockIdx.x * 256 + threadIdx.x;          // B*D threads
  int d = i % DD;
  int b = i / DD;
  float f = 1.f;
  #pragma unroll
  for (int k = 0; k < CL; ++k) f *= 0.9f;          // 0.9^64, constant-folded
  float e = 0.f;
  for (int c = 0; c < NC; ++c) {
    size_t idx = ((size_t)b * NC + c) * DD + d;
    estart[idx] = e;                               // e entering chunk c
    float S = carry[idx];
    e = (c == 0) ? S : f * e + S;
  }
}

__global__ __launch_bounds__(256) void ema_phaseC(const float* __restrict__ x,
                                                  const float* __restrict__ estart,
                                                  float* __restrict__ qtr,
                                                  float* __restrict__ del) {
  int i = blockIdx.x * 256 + threadIdx.x;
  int d = i % DD;
  int c = (i / DD) % NC;
  int b = i / (DD * NC);
  size_t base = ((size_t)b * TT + c * CL) * DD + d;
  float eprev = (c == 0) ? 0.f : estart[((size_t)b * NC + c) * DD + d];
  #pragma unroll 4
  for (int t = 0; t < CL; ++t) {
    int tt = c * CL + t;
    size_t a = base + (size_t)t * DD;
    float xv = x[a];
    float e = (tt == 0) ? xv : 0.9f * eprev + 0.1f * xv;
    qtr[a] = xv - e;
    del[a] = (tt == 0) ? 0.f : e - eprev;
    eprev = e;
  }
}

// =================================================================
// Tiled GEMM: out[M,N] = A[M,K](f32) * W[K,N](f32) + bias, bf16 WMMA
// block tile 128x128, 8 waves (4 x 2), wave tile 32x64 (2x4 accums)
// =================================================================
template <int OUT_BF16>
__global__ __launch_bounds__(256) void gemm_kernel(const float* __restrict__ A,
                                                   const float* __restrict__ W,
                                                   const float* __restrict__ bias,
                                                   void* __restrict__ out,
                                                   int M, int N, int K) {
  __shared__ __align__(16) unsigned short lX[128 * 32];  // [m][k]
  __shared__ __align__(16) unsigned short lW[128 * 32];  // [n][k] (transposed)
  const int tid = threadIdx.x;
  const int lane = tid & 31, wid = tid >> 5;
  const int g = lane >> 4, ln = lane & 15;
  const int wm = (wid & 3) * 32;       // 4 waves along M
  const int wn = (wid >> 2) * 64;      // 2 waves along N
  const int mBase = blockIdx.x * 128;
  const int nBase = blockIdx.y * 128;

  v8f acc[2][4] = {};

  for (int kt = 0; kt < K; kt += 32) {
    {  // stage X 128x32 (f32 -> bf16), packed 8B LDS stores
      int col = (tid & 7) * 4;
      int row0 = tid >> 3;
      #pragma unroll
      for (int p = 0; p < 4; ++p) {
        int row = row0 + p * 32;
        float4 f = *(const float4*)(A + (size_t)(mBase + row) * K + kt + col);
        u32x2 pk;
        pk.x = (unsigned)f2bf(f.x) | ((unsigned)f2bf(f.y) << 16);
        pk.y = (unsigned)f2bf(f.z) | ((unsigned)f2bf(f.w) << 16);
        *(u32x2*)&lX[row * 32 + col] = pk;
      }
    }
    {  // stage W 32x128 transposed into [n][k]: 4 strided loads -> 8B store
      int n = tid & 127;
      int kq = (tid >> 7) * 4;         // 0 or 4
      #pragma unroll
      for (int p = 0; p < 4; ++p) {
        int k = kq + p * 8;
        float f0 = W[(size_t)(kt + k + 0) * N + nBase + n];
        float f1 = W[(size_t)(kt + k + 1) * N + nBase + n];
        float f2 = W[(size_t)(kt + k + 2) * N + nBase + n];
        float f3 = W[(size_t)(kt + k + 3) * N + nBase + n];
        u32x2 pk;
        pk.x = (unsigned)f2bf(f0) | ((unsigned)f2bf(f1) << 16);
        pk.y = (unsigned)f2bf(f2) | ((unsigned)f2bf(f3) << 16);
        *(u32x2*)&lW[n * 32 + k] = pk;
      }
    }
    __syncthreads();

    Frag a[2], b[4];
    #pragma unroll
    for (int i = 0; i < 2; ++i) {  // A frag: lane = row, K pattern {8g..8g+7, 16+8g..}
      const unsigned short* p = &lX[(wm + i * 16 + ln) * 32 + g * 8];
      a[i].q[0] = *(const u32x4*)(p);
      a[i].q[1] = *(const u32x4*)(p + 16);
    }
    #pragma unroll
    for (int j = 0; j < 4; ++j) {  // B frag: lane = col, K pattern {8g..8g+7, 16+8g..}
      const unsigned short* p = &lW[(wn + j * 16 + ln) * 32 + g * 8];
      b[j].q[0] = *(const u32x4*)(p);
      b[j].q[1] = *(const u32x4*)(p + 16);
    }
    #pragma unroll
    for (int i = 0; i < 2; ++i)
      #pragma unroll
      for (int j = 0; j < 4; ++j)
        acc[i][j] = __builtin_amdgcn_wmma_f32_16x16x32_bf16(
            false, a[i].v, false, b[j].v, (short)0, acc[i][j], false, false);
    __syncthreads();
  }

  // epilogue: C layout -> vgpr r is row (r + 8g), col = ln
  #pragma unroll
  for (int i = 0; i < 2; ++i)
    #pragma unroll
    for (int j = 0; j < 4; ++j)
      #pragma unroll
      for (int r = 0; r < 8; ++r) {
        int m = mBase + wm + i * 16 + r + 8 * g;
        int n = nBase + wn + j * 16 + ln;
        float val = acc[i][j][r] + bias[n];
        if (OUT_BF16)
          ((unsigned short*)out)[(size_t)m * N + n] = f2bf(val);
        else
          ((float*)out)[(size_t)m * N + n] = val;
      }
}

// =================================================================
// Flash attention: Q bf16 [B*T, D], KV bf16 [B*T, 2D] (K | V), out f32
// 1 block = (b, h, 64 query rows); 4 waves x 16 rows; key tiles of 32
// =================================================================
__global__ __launch_bounds__(128) void attn_kernel(const unsigned short* __restrict__ Q,
                                                   const unsigned short* __restrict__ KV,
                                                   float* __restrict__ out) {
  __shared__ __align__(16) unsigned short lK[32 * 64];   // [key][hd]
  __shared__ __align__(16) unsigned short lVt[64 * 32];  // [hd][key]
  __shared__ __align__(16) unsigned short lP[4 * 16 * 32];
  const int tid = threadIdx.x;
  const int lane = tid & 31, wid = tid >> 5;
  const int g = lane >> 4, ln = lane & 15;
  const int nQB = TT / 64;
  const int qb = blockIdx.x % nQB;
  const int h = (blockIdx.x / nQB) % HH;
  const int b = blockIdx.x / (nQB * HH);
  const int kvld = 2 * DD;
  const size_t row0 = (size_t)b * TT;
  const int qbase = qb * 64 + wid * 16;

  Frag aq[2];  // Q fragments for hd chunks [0,32) and [32,64)
  {
    const unsigned short* qp = Q + (row0 + qbase + ln) * DD + h * HDIM;
    #pragma unroll
    for (int c = 0; c < 2; ++c) {
      aq[c].q[0] = *(const u32x4*)(qp + c * 32 + g * 8);
      aq[c].q[1] = *(const u32x4*)(qp + c * 32 + 16 + g * 8);
    }
  }

  float m_s[8], l_s[8];
  #pragma unroll
  for (int r = 0; r < 8; ++r) { m_s[r] = -1e30f; l_s[r] = 0.f; }
  v8f o[4] = {};
  const float sc = 0.125f;  // HD^-0.5
  unsigned short* lPw = &lP[wid * 16 * 32];

  for (int kt = 0; kt < TT; kt += 32) {
    {  // stage K tile [key][hd]: async global->LDS when available
      int col8 = (tid & 7) * 8;
      int r0 = tid >> 3;
      #pragma unroll
      for (int p = 0; p < 2; ++p) {
        int kr = r0 + p * 16;
        const unsigned short* ks =
            KV + (row0 + kt + kr) * (size_t)kvld + h * HDIM + col8;
#if USE_ASYNC_LDS
        async_cp16(ks, &lK[kr * 64 + col8]);
#else
        *(u32x4*)&lK[kr * 64 + col8] = *(const u32x4*)ks;
#endif
      }
    }
    {  // stage V^T tile [hd][key]: 4 strided loads -> one 8B LDS store
      int hd = tid & 63;
      int kq = (tid >> 6) * 4;   // 0 or 4
      const unsigned short* vbase =
          KV + (row0 + kt) * (size_t)kvld + DD + h * HDIM + hd;
      #pragma unroll
      for (int p = 0; p < 4; ++p) {
        int k = kq + p * 8;
        unsigned short e0 = vbase[(size_t)(k + 0) * kvld];
        unsigned short e1 = vbase[(size_t)(k + 1) * kvld];
        unsigned short e2 = vbase[(size_t)(k + 2) * kvld];
        unsigned short e3 = vbase[(size_t)(k + 3) * kvld];
        u32x2 pk;
        pk.x = (unsigned)e0 | ((unsigned)e1 << 16);
        pk.y = (unsigned)e2 | ((unsigned)e3 << 16);
        *(u32x2*)&lVt[hd * 32 + k] = pk;
      }
    }
#if USE_ASYNC_LDS
    __builtin_amdgcn_s_wait_asynccnt(0);
#endif
    __syncthreads();

    // S = Q K^T : two 16-key subtiles, K-dim split into two 32-chunks of hd
    v8f s[2] = {};
    #pragma unroll
    for (int sub = 0; sub < 2; ++sub) {
      const unsigned short* kb = &lK[(sub * 16 + ln) * 64];
      #pragma unroll
      for (int c = 0; c < 2; ++c) {
        Frag bk;
        bk.q[0] = *(const u32x4*)(kb + c * 32 + g * 8);
        bk.q[1] = *(const u32x4*)(kb + c * 32 + g * 8 + 16);
        s[sub] = __builtin_amdgcn_wmma_f32_16x16x32_bf16(
            false, aq[c].v, false, bk.v, (short)0, s[sub], false, false);
      }
    }

    // online softmax with DPP row reductions; P -> per-wave LDS as bf16
    #pragma unroll
    for (int r = 0; r < 8; ++r) {
      float a0 = s[0][r] * sc, a1 = s[1][r] * sc;
      float lm = redmax16(fmaxf(a0, a1));
      float mnew = fmaxf(m_s[r], lm);
      float alpha = __expf(m_s[r] - mnew);
      float p0 = __expf(a0 - mnew);
      float p1 = __expf(a1 - mnew);
      float rs = redsum16(p0 + p1);
      l_s[r] = l_s[r] * alpha + rs;
      m_s[r] = mnew;
      #pragma unroll
      for (int j = 0; j < 4; ++j) o[j][r] *= alpha;
      lPw[(r + 8 * g) * 32 + ln] = f2bf(p0);
      lPw[(r + 8 * g) * 32 + 16 + ln] = f2bf(p1);
    }

    Frag pf;  // A fragment of P (16 rows x 32 keys)
    {
      const unsigned short* pb = &lPw[ln * 32 + g * 8];
      pf.q[0] = *(const u32x4*)(pb);
      pf.q[1] = *(const u32x4*)(pb + 16);
    }
    #pragma unroll
    for (int j = 0; j < 4; ++j) {  // O += P * V, 4 hd chunks of 16
      const unsigned short* vb = &lVt[(j * 16 + ln) * 32 + g * 8];
      Frag vf;
      vf.q[0] = *(const u32x4*)(vb);
      vf.q[1] = *(const u32x4*)(vb + 16);
      o[j] = __builtin_amdgcn_wmma_f32_16x16x32_bf16(
          false, pf.v, false, vf.v, (short)0, o[j], false, false);
    }
    __syncthreads();
  }

  #pragma unroll
  for (int r = 0; r < 8; ++r) {
    float inv = 1.0f / l_s[r];
    int qrow = qbase + r + 8 * g;
    float* op = out + (row0 + qrow) * (size_t)DD + h * HDIM + ln;
    #pragma unroll
    for (int j = 0; j < 4; ++j) op[j * 16] = o[j][r] * inv;
  }
}

// =================================================================
// gated fusion: gs = (1-w)*gs + w*gc, w = sigmoid(fusion_weight)
// =================================================================
__global__ __launch_bounds__(256) void fuse_kernel(float* __restrict__ gs,
                                                   const float* __restrict__ gc,
                                                   const float* __restrict__ fw) {
  float w = 1.f / (1.f + __expf(-fw[0]));
  size_t i = ((size_t)blockIdx.x * 256 + threadIdx.x) * 4;
  float4 a = *(float4*)(gs + i);
  float4 c = *(const float4*)(gc + i);
  a.x = (1.f - w) * a.x + w * c.x;
  a.y = (1.f - w) * a.y + w * c.y;
  a.z = (1.f - w) * a.z + w * c.z;
  a.w = (1.f - w) * a.w + w * c.w;
  *(float4*)(gs + i) = a;
}

// =================================================================
extern "C" void kernel_launch(void* const* d_in, const int* in_sizes, int n_in,
                              void* d_out, int out_size, void* d_ws, size_t ws_size,
                              hipStream_t stream) {
  const float* x        = (const float*)d_in[0];
  const float* Wq_self  = (const float*)d_in[1];
  const float* bq_self  = (const float*)d_in[2];
  const float* Wkv_self = (const float*)d_in[3];
  const float* bkv_self = (const float*)d_in[4];
  const float* Wq_cross = (const float*)d_in[5];
  const float* bq_cross = (const float*)d_in[6];
  const float* Wkv_ema  = (const float*)d_in[7];
  const float* bkv_ema  = (const float*)d_in[8];
  const float* Wproj    = (const float*)d_in[9];
  const float* bproj    = (const float*)d_in[10];
  const float* fusion_w = (const float*)d_in[11];

  const size_t MB = 1024 * 1024;
  char* ws = (char*)d_ws;
  float*          qtr   = (float*)(ws + 0 * MB);            // 16 MB, later reused as out_self
  float*          del   = (float*)(ws + 16 * MB);           // 16 MB, later reused as out_cross
  unsigned short* q_s   = (unsigned short*)(ws + 32 * MB);  //  8 MB bf16
  unsigned short* kv_s  = (unsigned short*)(ws + 40 * MB);  // 16 MB bf16
  unsigned short* q_c   = (unsigned short*)(ws + 56 * MB);  //  8 MB bf16
  unsigned short* kv_c  = (unsigned short*)(ws + 64 * MB);  // 16 MB bf16
  float*          carry = (float*)(ws + 80 * MB);           // 256 KB
  float*          estart = carry + (size_t)BB * NC * DD;    // 256 KB
  float* out_self  = qtr;
  float* out_cross = del;

  // EMA scan -> q_trend, delta
  ema_phaseA<<<(BB * NC * DD) / 256, 256, 0, stream>>>(x, carry);
  ema_phaseB<<<(BB * DD) / 256, 256, 0, stream>>>(carry, estart);
  ema_phaseC<<<(BB * NC * DD) / 256, 256, 0, stream>>>(x, estart, qtr, del);

  dim3 blk(256);
  // projections (bf16 outputs feed WMMA attention)
  gemm_kernel<1><<<dim3(BT / 128, DD / 128), blk, 0, stream>>>(x, Wq_self, bq_self, q_s, BT, DD, DD);
  gemm_kernel<1><<<dim3(BT / 128, (2 * DD) / 128), blk, 0, stream>>>(x, Wkv_self, bkv_self, kv_s, BT, 2 * DD, DD);
  gemm_kernel<1><<<dim3(BT / 128, DD / 128), blk, 0, stream>>>(qtr, Wq_cross, bq_cross, q_c, BT, DD, DD);
  gemm_kernel<1><<<dim3(BT / 128, (2 * DD) / 128), blk, 0, stream>>>(del, Wkv_ema, bkv_ema, kv_c, BT, 2 * DD, DD);

  // attention branches (overwrite qtr/del regions, which are now dead)
  attn_kernel<<<BB * HH * (TT / 64), 128, 0, stream>>>(q_s, kv_s, out_self);
  attn_kernel<<<BB * HH * (TT / 64), 128, 0, stream>>>(q_c, kv_c, out_cross);

  // gated fusion (in place into out_self)
  fuse_kernel<<<(int)(BTD / 1024), 256, 0, stream>>>(out_self, out_cross, fusion_w);

  // output projection (f32 output)
  gemm_kernel<0><<<dim3(BT / 128, DD / 128), blk, 0, stream>>>(out_self, Wproj, bproj, (float*)d_out, BT, DD, DD);
}
